// LSHAttnBucketing_37383395344518
// MI455X (gfx1250) — matrix operand order repaired and checked by reading, not compiled, not run on hardware
//
#include <hip/hip_runtime.h>
#include <hip/hip_bf16.h>

typedef __attribute__((ext_vector_type(16))) _Float16 v16h;
typedef __attribute__((ext_vector_type(8)))  _Float16 v8h;
typedef __attribute__((ext_vector_type(8)))  float    v8f;

#define NPTS 32768
#define HDIM 64
#define NH   8
#define DH   64
#define LR   3
#define BSZ  128
#define NBK  256
#define EPAD 128   // E = 104 padded to 128 (multiple of K=32)
#define LH   (LR*NH)

// ---------------------------------------------------------------------------
// WMMA helpers (gfx1250 wave32, V_WMMA_F32_16X16X32_F16)
// A-frag layout (16x32 f16): lanes 0-15 -> M=lane, K runs [0..7],[16..23];
// lanes 16-31 -> M=lane-16, K runs [8..15],[24..31]. Two 16B loads per lane.
// B-frag layout (32x16 f16): lane holds column n = n0+lane%16, K runs
// [kb..kb+7],[kb+16..kb+23] with kb = k0 + 8*(lane/16). Two 16B loads per lane.
// C/D: v8f, row = r + 8*(lane/16), col = lane%16.
// ---------------------------------------------------------------------------
__device__ __forceinline__ v16h load_a16(const _Float16* base, int pitch, int m0, int k0) {
    int lane = threadIdx.x & 31;
    int m  = m0 + (lane & 15);
    int hh = lane >> 4;
    const _Float16* rp = base + (size_t)m * pitch + k0 + hh * 8;
    v8h lo = *(const v8h*)rp;
    v8h hi = *(const v8h*)(rp + 16);
    v16h r;
#pragma unroll
    for (int i = 0; i < 8; ++i) { r[i] = lo[i]; r[i + 8] = hi[i]; }
    return r;
}

__device__ __forceinline__ v16h load_b16(const _Float16* base, int pitch, int n0, int k0) {
    int lane = threadIdx.x & 31;
    int n  = n0 + (lane & 15);
    int kb = k0 + (lane >> 4) * 8;
    const _Float16* rp = base + (size_t)n * pitch + kb;
    v8h lo = *(const v8h*)rp;
    v8h hi = *(const v8h*)(rp + 16);
    v16h r;
#pragma unroll
    for (int i = 0; i < 8; ++i) { r[i] = lo[i]; r[i + 8] = hi[i]; }
    return r;
}

__device__ __forceinline__ v8f wmma16(v16h a, v16h b, v8f c) {
    return __builtin_amdgcn_wmma_f32_16x16x32_f16(false, a, false, b, (short)0, c, false, false);
}

// ---------------------------------------------------------------------------
// f32 -> f16 conversion
// ---------------------------------------------------------------------------
__global__ void k_cvt(const float* __restrict__ src, _Float16* __restrict__ dst, int cnt) {
    int i = blockIdx.x * blockDim.x + threadIdx.x;
    if (i < cnt) dst[i] = (_Float16)src[i];
}

// ---------------------------------------------------------------------------
// RPE distance weights: sw[h][j] = sqrt(2*|mean_d W[h,d,j]| + 1e-12), j<40
// ---------------------------------------------------------------------------
__global__ void k_sw(const float* __restrict__ wrpe, float* __restrict__ swb) {
    int i = blockIdx.x * blockDim.x + threadIdx.x;
    if (i >= NH * 40) return;
    int h = i / 40, j = i % 40;
    float s = 0.f;
    for (int d = 0; d < DH; ++d) s += wrpe[(size_t)(h * DH + d) * 40 + j];
    s = fabsf(s * (1.f / DH));
    swb[i] = sqrtf(2.f * s + 1e-12f);
}

// ---------------------------------------------------------------------------
// LayerNorm -> f16
// ---------------------------------------------------------------------------
__global__ void k_ln(const float* __restrict__ x, const float* __restrict__ g,
                     const float* __restrict__ b, _Float16* __restrict__ xn) {
    int n = blockIdx.x * blockDim.x + threadIdx.x;
    if (n >= NPTS) return;
    const float* xr = x + (size_t)n * HDIM;
    float mu = 0.f;
    for (int d = 0; d < HDIM; ++d) mu += xr[d];
    mu *= (1.f / HDIM);
    float var = 0.f;
    for (int d = 0; d < HDIM; ++d) { float t = xr[d] - mu; var += t * t; }
    var *= (1.f / HDIM);
    float rs = rsqrtf(var + 1e-5f);
    for (int d = 0; d < HDIM; ++d)
        xn[(size_t)n * HDIM + d] = (_Float16)((xr[d] - mu) * rs * g[d] + b[d]);
}

// ---------------------------------------------------------------------------
// Fused QKV projection: out(N,1536) = xn(N,64) @ Wf(1536,64)^T via WMMA.
// One wave per 16x16 tile. Region select is UNIFORM (on c0), so the store
// tail is scalar-branched straight-line code.
// ---------------------------------------------------------------------------
__global__ void k_qkv(const _Float16* __restrict__ xn, const _Float16* __restrict__ Wf,
                      _Float16* __restrict__ qhat, _Float16* __restrict__ khat,
                      _Float16* __restrict__ vbuf) {
    int m0 = blockIdx.x * 16;
    int c0 = blockIdx.y * 16;
    int lane = threadIdx.x & 31;
    v8f acc = {};
#pragma unroll
    for (int ks = 0; ks < 2; ++ks) {
        v16h a = load_a16(xn, HDIM, m0, ks * 32);
        v16h b = load_b16(Wf, HDIM, c0, ks * 32);
        acc = wmma16(a, b, acc);
    }
    int hh = lane >> 4, nl = lane & 15;
    if (c0 < 512) {
        int c = c0 + nl;
#pragma unroll
        for (int r = 0; r < 8; ++r) {
            int row = m0 + r + 8 * hh;
            qhat[((size_t)row * NH + (c >> 6)) * EPAD + (c & 63)] = (_Float16)acc[r];
        }
    } else if (c0 < 1024) {
        int c = c0 - 512 + nl;
#pragma unroll
        for (int r = 0; r < 8; ++r) {
            int row = m0 + r + 8 * hh;
            khat[((size_t)row * NH + (c >> 6)) * EPAD + (c & 63)] = (_Float16)acc[r];
        }
    } else {
        int c = c0 - 1024 + nl;
#pragma unroll
        for (int r = 0; r < 8; ++r) {
            int row = m0 + r + 8 * hh;
            vbuf[(size_t)row * (NH * DH) + c] = (_Float16)acc[r];
        }
    }
}

// ---------------------------------------------------------------------------
// Build RPE tail of q_hat/k_hat, norms, and E2LSH hash values. One thread/(n,h).
// ---------------------------------------------------------------------------
__global__ void k_hash(_Float16* __restrict__ qhat, _Float16* __restrict__ khat,
                       const float* __restrict__ coords, const float* __restrict__ swb,
                       const float* __restrict__ alpha, const float* __restrict__ beta,
                       float* __restrict__ qnorm, float* __restrict__ knorm,
                       float* __restrict__ hq, float* __restrict__ hk) {
    int idx = blockIdx.x * blockDim.x + threadIdx.x;
    if (idx >= NPTS * NH) return;
    int n = idx >> 3, h = idx & 7;
    _Float16* qr = qhat + (size_t)idx * EPAD;
    _Float16* kr = khat + (size_t)idx * EPAD;
    float qv[64], kv[64];
    float qn2 = 0.f, kn2 = 0.f;
    for (int e = 0; e < 64; ++e) {
        qv[e] = (float)qr[e]; kv[e] = (float)kr[e];
        qn2 += qv[e] * qv[e]; kn2 += kv[e] * kv[e];
    }
    float cd[6];
    for (int c = 0; c < 6; ++c) cd[c] = coords[(size_t)n * 6 + c];
    float rpn = 0.f;
    for (int j = 0; j < 40; ++j) {
        float rp = swb[h * 40 + j] * cd[j % 5];
        rpn += rp * rp;
        _Float16 rh = (_Float16)rp;
        qr[64 + j] = rh; kr[64 + j] = rh;
    }
    for (int j = 104; j < 128; ++j) { qr[j] = (_Float16)0.f; kr[j] = (_Float16)0.f; }
    qnorm[idx] = -0.5f * (qn2 + rpn);
    knorm[idx] = -0.5f * (kn2 + rpn);
    for (int l = 0; l < LR; ++l) {
        const float* al = alpha + ((size_t)(l * NH + h)) * 70;
        float bb = beta[l * NH + h];
        float hqv = bb, hkv = bb;
        for (int e = 0; e < 64; ++e) { hqv += al[e] * qv[e]; hkv += al[e] * kv[e]; }
        for (int c = 0; c < 6; ++c) { float a6 = al[64 + c]; hqv += a6 * cd[c]; hkv += a6 * cd[c]; }
        hq[((size_t)(l * NH + h)) * NPTS + n] = hqv;
        hk[((size_t)(l * NH + h)) * NPTS + n] = hkv;
    }
}

// ---------------------------------------------------------------------------
// Full 32768-element argsort per (l,h,{q,k}) in ONE workgroup's 256KB LDS
// (CDNA5: 320KB/WGP). (key|idx) packed into sortable u64, bitonic sort.
// ---------------------------------------------------------------------------
__global__ void k_sort(const float* __restrict__ hq, const float* __restrict__ hk,
                       unsigned* __restrict__ permq, unsigned* __restrict__ permk) {
    extern __shared__ char smem[];
    unsigned long long* s = (unsigned long long*)smem;
    int lh = blockIdx.x;
    const float* kp = (blockIdx.y == 0 ? hq : hk) + (size_t)lh * NPTS;
    unsigned* pp = (blockIdx.y == 0 ? permq : permk) + (size_t)lh * NPTS;
    int tid = threadIdx.x;
    for (int i = tid; i < NPTS; i += 1024) {
        unsigned u = __float_as_uint(kp[i]);
        u = (u & 0x80000000u) ? ~u : (u | 0x80000000u);   // monotonic float->uint
        s[i] = ((unsigned long long)u << 32) | (unsigned)i;
    }
    __syncthreads();
    for (int k = 2; k <= NPTS; k <<= 1) {
        for (int j = k >> 1; j > 0; j >>= 1) {
            for (int i = tid; i < NPTS; i += 1024) {
                int ixj = i ^ j;
                if (ixj > i) {
                    unsigned long long a = s[i], c = s[ixj];
                    bool up = ((i & k) == 0);
                    if (up ? (a > c) : (a < c)) { s[i] = c; s[ixj] = a; }
                }
            }
            __syncthreads();
        }
    }
    for (int i = tid; i < NPTS; i += 1024) pp[i] = (unsigned)(s[i] & 0xFFFFFFFFu);
}

// ---------------------------------------------------------------------------
// Bucketed kernel-attention. One workgroup (8 wave32) per (bucket, l, h).
// LDS: sq 32K | sk 32K | svT 16K | P 32K | qn/kn/orig 2K  (~114KB)
// Each wave: 16x128 S-strip. A-frags of sq are hoisted into registers and
// reused across all 8 column tiles; both tile loops fully unrolled so the
// scheduler can overlap the WMMA chains with v_exp_f32 work.
// ---------------------------------------------------------------------------
__global__ void __launch_bounds__(256) k_attn(
        const _Float16* __restrict__ qhat, const _Float16* __restrict__ khat,
        const _Float16* __restrict__ vbuf,
        const float* __restrict__ qnorm, const float* __restrict__ knorm,
        const unsigned* __restrict__ permq, const unsigned* __restrict__ permk,
        _Float16* __restrict__ numL, float* __restrict__ denomL) {
    extern __shared__ char smem[];
    _Float16* s_sq  = (_Float16*)(smem);                 // 128 x 128
    _Float16* s_sk  = (_Float16*)(smem + 32768);         // 128 x 128
    _Float16* s_svT = (_Float16*)(smem + 65536);         // 64  x 128 (transposed V)
    _Float16* s_P   = (_Float16*)(smem + 81920);         // 8 waves x 16 x 128
    float*    s_qn  = (float*)(smem + 114688);           // 128
    float*    s_kn  = (float*)(smem + 115200);           // 128
    unsigned* s_oq  = (unsigned*)(smem + 115712);        // 128
    unsigned* s_ok  = (unsigned*)(smem + 116224);        // 128

    int tid = threadIdx.x;
    int b = blockIdx.x;
    int lh = blockIdx.y;
    int l = lh >> 3, h = lh & 7;
    size_t lhN = (size_t)lh * NPTS;

    for (int i = tid; i < BSZ; i += 256) {
        unsigned oq = permq[lhN + (size_t)b * BSZ + i];
        unsigned ok = permk[lhN + (size_t)b * BSZ + i];
        s_oq[i] = oq; s_ok[i] = ok;
        s_qn[i] = qnorm[(size_t)oq * NH + h];
        s_kn[i] = knorm[(size_t)ok * NH + h];
    }
    __syncthreads();
    // gather sorted q_hat / k_hat rows (128 halfs = 64 dwords each)
    for (int idx = tid; idx < BSZ * 64; idx += 256) {
        int row = idx >> 6, dw = idx & 63;
        ((unsigned*)s_sq)[idx] =
            ((const unsigned*)(qhat + ((size_t)s_oq[row] * NH + h) * EPAD))[dw];
        ((unsigned*)s_sk)[idx] =
            ((const unsigned*)(khat + ((size_t)s_ok[row] * NH + h) * EPAD))[dw];
    }
    // gather V transposed: svT[d][j] = v[perm_k[j]][h][d]
    for (int idx = tid; idx < BSZ * DH; idx += 256) {
        int j = idx >> 6, d = idx & 63;
        s_svT[d * BSZ + j] = vbuf[(size_t)s_ok[j] * (NH * DH) + h * DH + d];
    }
    __syncthreads();

    int wave = tid >> 5, lane = tid & 31;
    int hh = lane >> 4, nl = lane & 15;
    int m0 = wave * 16;
    _Float16* Pw = s_P + (size_t)wave * 16 * EPAD;

    float qn_r[8], rowsum[8];
#pragma unroll
    for (int r = 0; r < 8; ++r) { qn_r[r] = s_qn[m0 + r + 8 * hh]; rowsum[r] = 0.f; }

    // hoist the 4 A-frags of this wave's 16-row strip of sq (K = 128 padded)
    v16h aq[4];
#pragma unroll
    for (int ks = 0; ks < 4; ++ks) aq[ks] = load_a16(s_sq, EPAD, m0, ks * 32);

    // S = sq . sk^T, exp kernel, stage P as f16
#pragma unroll
    for (int ct = 0; ct < 8; ++ct) {
        v8f acc = {};
#pragma unroll
        for (int ks = 0; ks < 4; ++ks)
            acc = wmma16(aq[ks], load_b16(s_sk, EPAD, ct * 16, ks * 32), acc);
        float knc = s_kn[ct * 16 + nl];
#pragma unroll
        for (int r = 0; r < 8; ++r) {
            float logit = acc[r] + qn_r[r] + knc;
            float val = __expf(fminf(logit, 0.f));
            rowsum[r] += val;
            Pw[(r + 8 * hh) * EPAD + ct * 16 + nl] = (_Float16)val;
        }
    }
    __syncthreads();

    // row sums: reduce over the 16 lanes of each half (column partials)
#pragma unroll
    for (int r = 0; r < 8; ++r) {
        float s = rowsum[r];
        s += __shfl_xor(s, 1, 32);
        s += __shfl_xor(s, 2, 32);
        s += __shfl_xor(s, 4, 32);
        s += __shfl_xor(s, 8, 32);
        rowsum[r] = s;
    }
    if (nl == 0) {
#pragma unroll
        for (int r = 0; r < 8; ++r) {
            unsigned orig = s_oq[m0 + r + 8 * hh];
            denomL[((size_t)l * NPTS + orig) * NH + h] = rowsum[r] + 1e-20f;
        }
    }

    // numerator = P(16x128) @ sv(128x64); scatter to original point order
    v16h ap[4];
#pragma unroll
    for (int ks = 0; ks < 4; ++ks) ap[ks] = load_a16(Pw, EPAD, 0, ks * 32);
#pragma unroll
    for (int ct = 0; ct < 4; ++ct) {
        v8f acc = {};
#pragma unroll
        for (int ks = 0; ks < 4; ++ks)
            acc = wmma16(ap[ks], load_b16(s_svT, BSZ, ct * 16, ks * 32), acc);
#pragma unroll
        for (int r = 0; r < 8; ++r) {
            unsigned orig = s_oq[m0 + r + 8 * hh];
            numL[((size_t)l * NPTS + orig) * (NH * DH) + h * DH + ct * 16 + nl] =
                (_Float16)acc[r];
        }
    }
}

// ---------------------------------------------------------------------------
// Combine rounds: out = (sum_l num_l) / (sum_l denom_l)  -> f16 for final GEMM
// ---------------------------------------------------------------------------
__global__ void k_combine(const _Float16* __restrict__ numL, const float* __restrict__ denomL,
                          _Float16* __restrict__ mixed) {
    size_t idx = (size_t)blockIdx.x * blockDim.x + threadIdx.x;
    int n = (int)(idx >> 9);
    int c = (int)(idx & 511);
    int h = c >> 6;
    float ds = 0.f, nm = 0.f;
#pragma unroll
    for (int l = 0; l < LR; ++l) {
        ds += denomL[((size_t)l * NPTS + n) * NH + h];
        nm += (float)numL[((size_t)l * NPTS + n) * (NH * DH) + c];
    }
    mixed[idx] = (_Float16)(nm / ds);
}

// ---------------------------------------------------------------------------
// Output projection: out(N,512) = mixed(N,512) @ wo(512,512)^T + b (WMMA, K=512)
// ---------------------------------------------------------------------------
__global__ void k_out(const _Float16* __restrict__ mixed, const _Float16* __restrict__ woF,
                      const float* __restrict__ wob, float* __restrict__ out) {
    int m0 = blockIdx.x * 16;
    int o0 = blockIdx.y * 16;
    int lane = threadIdx.x & 31;
    v8f acc = {};
#pragma unroll 4
    for (int ks = 0; ks < 16; ++ks) {
        v16h a = load_a16(mixed, 512, m0, ks * 32);
        v16h b = load_b16(woF, 512, o0, ks * 32);
        acc = wmma16(a, b, acc);
    }
    int hh = lane >> 4, nl = lane & 15;
    float bias = wob[o0 + nl];
#pragma unroll
    for (int r = 0; r < 8; ++r)
        out[(size_t)(m0 + r + 8 * hh) * 512 + o0 + nl] = acc[r] + bias;
}

// ---------------------------------------------------------------------------
extern "C" void kernel_launch(void* const* d_in, const int* in_sizes, int n_in,
                              void* d_out, int out_size, void* d_ws, size_t ws_size,
                              hipStream_t stream) {
    (void)in_sizes; (void)n_in; (void)out_size; (void)ws_size;
    const float* x      = (const float*)d_in[0];
    const float* coords = (const float*)d_in[1];
    const float* Wq     = (const float*)d_in[2];
    const float* Wk     = (const float*)d_in[3];
    const float* Wv     = (const float*)d_in[4];
    const float* ln_g   = (const float*)d_in[5];
    const float* ln_b   = (const float*)d_in[6];
    const float* wrpe   = (const float*)d_in[7];
    const float* woW    = (const float*)d_in[8];
    const float* wob    = (const float*)d_in[9];
    const float* alpha  = (const float*)d_in[10];
    const float* beta   = (const float*)d_in[11];
    float* out = (float*)d_out;

    char* p = (char*)d_ws;
    auto carve = [&](size_t bytes) -> char* {
        char* r = p;
        p += (bytes + 255) & ~(size_t)255;
        return r;
    };
    _Float16* xn    = (_Float16*)carve((size_t)NPTS * HDIM * 2);
    _Float16* Wf    = (_Float16*)carve((size_t)1536 * HDIM * 2);
    _Float16* woF   = (_Float16*)carve((size_t)512 * 512 * 2);
    float*    swb   = (float*)carve((size_t)NH * 40 * 4);
    _Float16* qhat  = (_Float16*)carve((size_t)NPTS * NH * EPAD * 2);
    _Float16* khat  = (_Float16*)carve((size_t)NPTS * NH * EPAD * 2);
    _Float16* vbuf  = (_Float16*)carve((size_t)NPTS * NH * DH * 2);
    float*    qn    = (float*)carve((size_t)NPTS * NH * 4);
    float*    kn    = (float*)carve((size_t)NPTS * NH * 4);
    float*    hqb   = (float*)carve((size_t)LH * NPTS * 4);
    float*    hkb   = (float*)carve((size_t)LH * NPTS * 4);
    unsigned* permq = (unsigned*)carve((size_t)LH * NPTS * 4);
    unsigned* permk = (unsigned*)carve((size_t)LH * NPTS * 4);
    _Float16* numL  = (_Float16*)carve((size_t)LR * NPTS * NH * DH * 2);
    float*    denL  = (float*)carve((size_t)LR * NPTS * NH * 4);
    _Float16* mixed = (_Float16*)carve((size_t)NPTS * NH * DH * 2);

    // weight conversions to f16
    k_cvt<<<(512 * 64 + 255) / 256, 256, 0, stream>>>(Wq, Wf, 512 * 64);
    k_cvt<<<(512 * 64 + 255) / 256, 256, 0, stream>>>(Wk, Wf + 512 * 64, 512 * 64);
    k_cvt<<<(512 * 64 + 255) / 256, 256, 0, stream>>>(Wv, Wf + 1024 * 64, 512 * 64);
    k_cvt<<<(512 * 512 + 255) / 256, 256, 0, stream>>>(woW, woF, 512 * 512);
    k_sw<<<2, 256, 0, stream>>>(wrpe, swb);

    // layernorm + fused QKV projection (WMMA)
    k_ln<<<NPTS / 256, 256, 0, stream>>>(x, ln_g, ln_b, xn);
    k_qkv<<<dim3(NPTS / 16, 1536 / 16), 32, 0, stream>>>(xn, Wf, qhat, khat, vbuf);

    // RPE tail, norms, E2LSH hashes
    k_hash<<<(NPTS * NH) / 256, 256, 0, stream>>>(qhat, khat, coords, swb, alpha, beta,
                                                  qn, kn, hqb, hkb);

    // 48 full argsorts, each in one workgroup's 256KB LDS
    k_sort<<<dim3(LH, 2), 1024, (size_t)NPTS * 8, stream>>>(hqb, hkb, permq, permk);

    // bucketed attention (WMMA), scatter numerators/denominators
    k_attn<<<dim3(NBK, LH), 256, 116736, stream>>>(qhat, khat, vbuf, qn, kn,
                                                   permq, permk, numL, denL);

    // combine hash rounds + output projection (WMMA)
    k_combine<<<(NPTS * NH * DH) / 256, 256, 0, stream>>>(numL, denL, mixed);
    k_out<<<dim3(NPTS / 16, 512 / 16), 32, 0, stream>>>(mixed, woF, wob, out);
}
